// MeshDiT_6975026889225
// MI455X (gfx1250) — 16-bit and 8-bit matrix operand order repaired and checked by reading, compiled, not run on hardware
//
#include <hip/hip_runtime.h>
#include <cstdint>
#include <cstddef>


typedef __attribute__((ext_vector_type(2)))  int      v2i;
typedef __attribute__((ext_vector_type(4)))  int      v4i;
typedef __attribute__((ext_vector_type(8)))  int      v8i;
typedef __attribute__((ext_vector_type(8)))  float    v8f;
typedef __attribute__((ext_vector_type(8)))  _Float16 v8h;
typedef __attribute__((ext_vector_type(16))) _Float16 v16h;

#define HNUM 8
#define DDIM 512
#define HD   64

__device__ inline v8f wmma_f16f32(v16h a, v16h b, v8f c) {
  return __builtin_amdgcn_wmma_f32_16x16x32_f16(false, a, false, b, (short)0, c,
                                                false, false);
}

__global__ __launch_bounds__(256) void w_absmean(const float* __restrict__ W,
                                                 float* __restrict__ meanout, int n) {
  __shared__ float red[256];
  float s = 0.f;
  for (int i = threadIdx.x; i < n; i += 256) s += fabsf(W[i]);
  red[threadIdx.x] = s;
  __syncthreads();
  for (int st = 128; st > 0; st >>= 1) {
    if (threadIdx.x < st) red[threadIdx.x] += red[threadIdx.x + st];
    __syncthreads();
  }
  if (threadIdx.x == 0) meanout[0] = fmaxf(red[0] / (float)n, 1e-5f);
}

__global__ __launch_bounds__(256) void w_ternarize(const float* __restrict__ W,
                                                   const float* __restrict__ meanp,
                                                   signed char* __restrict__ Wq, int n) {
  int i = blockIdx.x * 256 + threadIdx.x;
  if (i >= n) return;
  float sc = 1.f / meanp[0];
  int q = __float2int_rn(W[i] * sc);
  q = q > 1 ? 1 : (q < -1 ? -1 : q);
  Wq[i] = (signed char)q;
}

__global__ __launch_bounds__(256) void rmsnorm_quant(const float* __restrict__ X,
                                                     const float* __restrict__ g,
                                                     signed char* __restrict__ Xq,
                                                     float* __restrict__ inv) {
  int t = blockIdx.x, tid = threadIdx.x;
  const float* xr = X + (size_t)t * DDIM;
  __shared__ float red[256];
  float x0 = xr[tid], x1 = xr[tid + 256];
  red[tid] = x0 * x0 + x1 * x1;
  __syncthreads();
  for (int st = 128; st > 0; st >>= 1) {
    if (tid < st) red[tid] += red[tid + st];
    __syncthreads();
  }
  float rinv = rsqrtf(red[0] * (1.f / (float)DDIM) + 1e-8f);
  __syncthreads();
  float n0 = x0 * rinv * g[tid], n1 = x1 * rinv * g[tid + 256];
  red[tid] = fmaxf(fabsf(n0), fabsf(n1));
  __syncthreads();
  for (int st = 128; st > 0; st >>= 1) {
    if (tid < st) red[tid] = fmaxf(red[tid], red[tid + st]);
    __syncthreads();
  }
  float mx = fmaxf(red[0], 1e-5f);
  float scale = 127.f / mx;
  int q0 = __float2int_rn(n0 * scale);
  int q1 = __float2int_rn(n1 * scale);
  q0 = q0 > 127 ? 127 : (q0 < -128 ? -128 : q0);
  q1 = q1 > 127 ? 127 : (q1 < -128 ? -128 : q1);
  Xq[(size_t)t * DDIM + tid] = (signed char)q0;
  Xq[(size_t)t * DDIM + tid + 256] = (signed char)q1;
  if (tid == 0) inv[t] = mx * (1.f / 127.f);
}

__global__ __launch_bounds__(256) void lowrank_t(const float* __restrict__ X,
                                                 const float* __restrict__ A,
                                                 float* __restrict__ T) {
  int m = blockIdx.x, tid = threadIdx.x;
  int r = tid & 15, seg = tid >> 4;
  const float* xr = X + (size_t)m * DDIM;
  const float* ar = A + (size_t)r * DDIM;
  float p = 0.f;
#pragma unroll 8
  for (int d = seg * 32; d < seg * 32 + 32; ++d) p += xr[d] * ar[d];
  __shared__ float red[16][17];
  red[r][seg] = p;
  __syncthreads();
  if (tid < 16) {
    float s = 0.f;
#pragma unroll
    for (int k = 0; k < 16; ++k) s += red[tid][k];
    T[(size_t)m * 16 + tid] = s;
  }
}

__global__ __launch_bounds__(256) void gemm_i8_wmma(
    const signed char* __restrict__ Aq, const signed char* __restrict__ Wq,
    const float* __restrict__ ainv, const float* __restrict__ winvp,
    int M, int N, int K, int mode, int Tper,
    const float* __restrict__ resid, float* __restrict__ outf,
    const float* __restrict__ tlow, const float* __restrict__ Bm,
    const float* __restrict__ mag,
    _Float16* __restrict__ qout, _Float16* __restrict__ kout,
    _Float16* __restrict__ vtout) {
  int lane = threadIdx.x & 31;
  int wave = threadIdx.x >> 5;
  int ng = N >> 6;
  int tile = blockIdx.x * 8 + wave;
  if (tile >= (M >> 4) * ng) return;
  int tm = (tile / ng) << 4;
  int tn0 = (tile % ng) << 6;

  int ln15 = lane & 15;
  int hi = lane >> 4;
  const signed char* aptr = Aq + (size_t)(tm + ln15) * K + (hi << 4);
  const signed char* bp0 = Wq + (size_t)(tn0 + 0  + ln15) * K + (hi << 4);
  const signed char* bp1 = Wq + (size_t)(tn0 + 16 + ln15) * K + (hi << 4);
  const signed char* bp2 = Wq + (size_t)(tn0 + 32 + ln15) * K + (hi << 4);
  const signed char* bp3 = Wq + (size_t)(tn0 + 48 + ln15) * K + (hi << 4);

  v8i acc[4] = {{}, {}, {}, {}};
  for (int kk = 0; kk < K; kk += 64) {
    __builtin_prefetch((const void*)(aptr + kk + 256), 0, 1);
    v2i a0 = *(const v2i*)(aptr + kk +  0);
    v2i a1 = *(const v2i*)(aptr + kk +  8);
    v2i a2 = *(const v2i*)(aptr + kk + 32);
    v2i a3 = *(const v2i*)(aptr + kk + 40);
    v8i af;
    af[0] = a0.x; af[1] = a0.y; af[2] = a1.x; af[3] = a1.y;
    af[4] = a2.x; af[5] = a2.y; af[6] = a3.x; af[7] = a3.y;
    const signed char* bps[4] = {bp0, bp1, bp2, bp3};
#pragma unroll
    for (int s = 0; s < 4; ++s) {
      v4i b0 = *(const v4i*)(bps[s] + kk +  0);
      v4i b1 = *(const v4i*)(bps[s] + kk + 32);
      v8i bf;
      bf[0] = b0.x; bf[1] = b0.y; bf[2] = b0.z; bf[3] = b0.w;
      bf[4] = b1.x; bf[5] = b1.y; bf[6] = b1.z; bf[7] = b1.w;
      acc[s] = __builtin_amdgcn_wmma_i32_16x16x64_iu8(true, af, true, bf, acc[s],
                                                      false, false);
    }
  }

  float winv = winvp[0];
  int mbase = tm + (hi << 3);
#pragma unroll
  for (int s = 0; s < 4; ++s) {
    int n = tn0 + (s << 4) + ln15;
#pragma unroll
    for (int j = 0; j < 8; ++j) {
      int m = mbase + j;
      float val = (float)acc[s][j] * ainv[m] * winv;
      if (mode == 0) {
        outf[(size_t)m * N + n] = resid[(size_t)m * N + n] + val;
      } else if (mode == 1) {
        int h = n >> 6, dh = n & 63;
        const float* tl = tlow + (size_t)m * 16;
        const float* bm = Bm + (size_t)n * 16;
        float dsum = 0.f;
#pragma unroll
        for (int r = 0; r < 16; ++r) dsum += tl[r] * bm[r];
        val += mag[h] * dsum;
        int b = m / Tper, tq = m % Tper;
        qout[(((size_t)b * HNUM + h) * Tper + tq) * HD + dh] = (_Float16)val;
      } else {
        int b = m / Tper, tc = m % Tper;
        if (n < (N >> 1)) {
          int h = n >> 6, dh = n & 63;
          kout[(((size_t)b * HNUM + h) * Tper + tc) * HD + dh] = (_Float16)val;
        } else {
          int n2 = n - (N >> 1);
          int h = n2 >> 6, dh = n2 & 63;
          vtout[(((size_t)b * HNUM + h) * HD + dh) * (size_t)Tper + tc] = (_Float16)val;
        }
      }
    }
  }
}

__global__ __launch_bounds__(32) void flash_attn(const _Float16* __restrict__ Q,
                                                 const _Float16* __restrict__ Kt,
                                                 const _Float16* __restrict__ Vt,
                                                 float* __restrict__ O,
                                                 int Tq, int Tc) {
  __shared__ __align__(32) float    Sbuf[2][16][32];
  __shared__ __align__(32) _Float16 Pbuf[2][16][32];
  __shared__ float rsBuf[32];
  __shared__ float liBuf[32];

  int lane = threadIdx.x;
  int qgroups = Tq >> 5;
  int qg = blockIdx.x % qgroups;
  int bh = blockIdx.x / qgroups;
  int h = bh & (HNUM - 1), b = bh / HNUM;

  const _Float16* qbase = Q + ((size_t)bh * Tq + (size_t)qg * 32) * HD;
  const _Float16* kbase = Kt + (size_t)bh * Tc * HD;
  const _Float16* vbase = Vt + (size_t)bh * HD * Tc;

  int ln15 = lane & 15;
  int hi = lane >> 4;
  int ka = hi << 4;
  int kb = hi << 4;
  int smb = hi << 3;

  union Fr { v16h v; v8h h2[2]; };
  Fr qf[2][2];
#pragma unroll
  for (int t = 0; t < 2; ++t) {
    const _Float16* qb = qbase + (size_t)(t * 16 + ln15) * HD;
    qf[t][0].h2[0] = *(const v8h*)(qb + ka);
    qf[t][0].h2[1] = *(const v8h*)(qb + ka + 8);
    qf[t][1].h2[0] = *(const v8h*)(qb + 32 + ka);
    qf[t][1].h2[1] = *(const v8h*)(qb + 32 + ka + 8);
  }

  v8f o[2][4] = {{{}, {}, {}, {}}, {{}, {}, {}, {}}};
  float m_i = -1e30f, l_i = 0.f;

  for (int c0 = 0; c0 < Tc; c0 += 32) {
    const _Float16* k0 = kbase + (size_t)(c0 + ln15) * HD;
    const _Float16* k1 = kbase + (size_t)(c0 + 16 + ln15) * HD;
    v16h bk00 = *(const v16h*)(k0 + kb);
    v16h bk01 = *(const v16h*)(k0 + 32 + kb);
    v16h bk10 = *(const v16h*)(k1 + kb);
    v16h bk11 = *(const v16h*)(k1 + 32 + kb);

#pragma unroll
    for (int t = 0; t < 2; ++t) {
      v8f s0 = {}, s1 = {};
      s0 = wmma_f16f32(qf[t][0].v, bk00, s0);
      s0 = wmma_f16f32(qf[t][1].v, bk01, s0);
      s1 = wmma_f16f32(qf[t][0].v, bk10, s1);
      s1 = wmma_f16f32(qf[t][1].v, bk11, s1);
#pragma unroll
      for (int j = 0; j < 8; ++j) {
        Sbuf[t][smb + j][ln15] = s0[j];
        Sbuf[t][smb + j][16 + ln15] = s1[j];
      }
    }
    __syncthreads();

    {
      float sc[32];
      float mx = m_i;
#pragma unroll
      for (int jj = 0; jj < 32; ++jj) {
        float s = Sbuf[hi][ln15][jj] * 0.125f;
        sc[jj] = s;
        mx = fmaxf(mx, s);
      }
      float rs = __expf(m_i - mx);
      float sum = 0.f;
#pragma unroll
      for (int jj = 0; jj < 32; ++jj) {
        float p = __expf(sc[jj] - mx);
        sum += p;
        Pbuf[hi][ln15][jj] = (_Float16)p;
      }
      l_i = l_i * rs + sum;
      m_i = mx;
      rsBuf[lane] = rs;
    }
    __syncthreads();

#pragma unroll
    for (int t = 0; t < 2; ++t) {
      float rs8[8];
#pragma unroll
      for (int j = 0; j < 8; ++j) rs8[j] = rsBuf[t * 16 + smb + j];
#pragma unroll
      for (int i = 0; i < 4; ++i)
#pragma unroll
        for (int j = 0; j < 8; ++j) o[t][i][j] *= rs8[j];
    }

    Fr pf0, pf1;
    pf0.h2[0] = *(const v8h*)(&Pbuf[0][ln15][ka]);
    pf0.h2[1] = *(const v8h*)(&Pbuf[0][ln15][ka + 8]);
    pf1.h2[0] = *(const v8h*)(&Pbuf[1][ln15][ka]);
    pf1.h2[1] = *(const v8h*)(&Pbuf[1][ln15][ka + 8]);
#pragma unroll
    for (int i = 0; i < 4; ++i) {
      v16h bv = *(const v16h*)(vbase + (size_t)(i * 16 + ln15) * Tc + c0 + kb);
      o[0][i] = wmma_f16f32(pf0.v, bv, o[0][i]);
      o[1][i] = wmma_f16f32(pf1.v, bv, o[1][i]);
    }
  }

  liBuf[lane] = 1.f / l_i;
  __syncthreads();
  float* obase = O + ((size_t)b * Tq + (size_t)qg * 32) * DDIM + h * HD;
#pragma unroll
  for (int t = 0; t < 2; ++t) {
    float li8[8];
#pragma unroll
    for (int j = 0; j < 8; ++j) li8[j] = liBuf[t * 16 + smb + j];
#pragma unroll
    for (int j = 0; j < 8; ++j) {
      size_t ro = (size_t)(t * 16 + smb + j) * DDIM;
      obase[ro + 0  + ln15] = o[t][0][j] * li8[j];
      obase[ro + 16 + ln15] = o[t][1][j] * li8[j];
      obase[ro + 32 + ln15] = o[t][2][j] * li8[j];
      obase[ro + 48 + ln15] = o[t][3][j] * li8[j];
    }
  }
}

extern "C" void kernel_launch(void* const* d_in, const int* in_sizes, int n_in,
                              void* d_out, int out_size, void* d_ws, size_t ws_size,
                              hipStream_t stream) {
  (void)in_sizes; (void)n_in; (void)out_size; (void)ws_size;

  const int B = 4, Nx = 4096, Nc = 1024, D = DDIM;
  const int Mx = B * Nx, Mc = B * Nc;

  const float* x = (const float*)d_in[0];
  const float* c = (const float*)d_in[1];
  float* out = (float*)d_out;

  char* ws = (char*)d_ws;
  size_t off = 0;
  auto alloc = [&](size_t bytes) -> void* {
    off = (off + 255) & ~(size_t)255;
    void* p = ws + off;
    off += bytes;
    return p;
  };
  float* wmean = (float*)alloc(6 * sizeof(float));
  const int widx[6]   = {2, 4, 9, 11, 13, 18};
  const int wsizes[6] = {D * D, 2 * D * D, D * D, D * D, 2 * D * D, D * D};
  signed char* wq_t[6];
  for (int i = 0; i < 6; ++i) wq_t[i] = (signed char*)alloc((size_t)wsizes[i]);
  signed char* act  = (signed char*)alloc((size_t)Mx * D);
  float* ainv_b     = (float*)alloc((size_t)Mx * sizeof(float));
  float* tlow       = (float*)alloc((size_t)Mx * 16 * sizeof(float));
  _Float16* qf      = (_Float16*)alloc((size_t)Mx * D * sizeof(_Float16));
  _Float16* kf      = (_Float16*)alloc((size_t)Mx * D * sizeof(_Float16));
  _Float16* vtf     = (_Float16*)alloc((size_t)Mx * D * sizeof(_Float16));
  float* attn       = (float*)alloc((size_t)Mx * D * sizeof(float));

  for (int i = 0; i < 6; ++i) {
    const float* W = (const float*)d_in[widx[i]];
    w_absmean<<<1, 256, 0, stream>>>(W, wmean + i, wsizes[i]);
    w_ternarize<<<(wsizes[i] + 255) / 256, 256, 0, stream>>>(W, wmean + i,
                                                             wq_t[i], wsizes[i]);
  }

  auto run_pass = [&](const float* qin, int Mq, int Tq,
                      const float* ctx, int Mctx, int Tc,
                      const signed char* WqT, const float* wq_inv,
                      const signed char* WkvT, const float* wkv_inv,
                      const signed char* WpT, const float* wp_inv,
                      const float* gq, const float* gkv, const float* gp,
                      const float* Amat, const float* Bmat, const float* magp,
                      float* outp) {
    rmsnorm_quant<<<Mq, 256, 0, stream>>>(qin, gq, act, ainv_b);
    lowrank_t<<<Mq, 256, 0, stream>>>(qin, Amat, tlow);
    {
      int tiles = (Mq / 16) * (D / 64);
      gemm_i8_wmma<<<(tiles + 7) / 8, 256, 0, stream>>>(
          act, WqT, ainv_b, wq_inv, Mq, D, D, 1, Tq,
          nullptr, nullptr, tlow, Bmat, magp, qf, nullptr, nullptr);
    }
    rmsnorm_quant<<<Mctx, 256, 0, stream>>>(ctx, gkv, act, ainv_b);
    {
      int tiles = (Mctx / 16) * ((2 * D) / 64);
      gemm_i8_wmma<<<(tiles + 7) / 8, 256, 0, stream>>>(
          act, WkvT, ainv_b, wkv_inv, Mctx, 2 * D, D, 2, Tc,
          nullptr, nullptr, nullptr, nullptr, nullptr, nullptr, kf, vtf);
    }
    flash_attn<<<B * HNUM * (Tq / 32), 32, 0, stream>>>(qf, kf, vtf, attn, Tq, Tc);
    rmsnorm_quant<<<Mq, 256, 0, stream>>>(attn, gp, act, ainv_b);
    {
      int tiles = (Mq / 16) * (D / 64);
      gemm_i8_wmma<<<(tiles + 7) / 8, 256, 0, stream>>>(
          act, WpT, ainv_b, wp_inv, Mq, D, D, 0, Tq,
          qin, outp, nullptr, nullptr, nullptr, nullptr, nullptr, nullptr);
    }
  };

  run_pass(x, Mx, Nx, c, Mc, Nc,
           wq_t[0], wmean + 0, wq_t[1], wmean + 1, wq_t[2], wmean + 2,
           (const float*)d_in[3], (const float*)d_in[5], (const float*)d_in[10],
           (const float*)d_in[6], (const float*)d_in[7], (const float*)d_in[8],
           out);
  run_pass(c, Mc, Nc, x, Mx, Nx,
           wq_t[3], wmean + 3, wq_t[4], wmean + 4, wq_t[5], wmean + 5,
           (const float*)d_in[12], (const float*)d_in[14], (const float*)d_in[19],
           (const float*)d_in[15], (const float*)d_in[16], (const float*)d_in[17],
           out + (size_t)Mx * D);
}
